// GDML_15049565405284
// MI455X (gfx1250) — matrix operand order repaired and checked by reading, not compiled, not run on hardware
//
#include <hip/hip_runtime.h>
#include <hip/hip_bf16.h>

// ---------------- problem constants ----------------
#define B_     16
#define N_     3136
#define C_     576
#define HEADS_ 8
#define AG_    49
#define HD_    72
#define HW_    56
#define QKV_   1728   // q | k | v fused columns
#define CHUNK_ 256

typedef __bf16 bf16;
typedef __attribute__((ext_vector_type(16))) __bf16 bf16x16;
typedef __attribute__((ext_vector_type(8)))  float  f32x8;
typedef __attribute__((ext_vector_type(4)))  unsigned int u32x4;
typedef __attribute__((ext_vector_type(8)))  int i32x8;
typedef __attribute__((ext_vector_type(4)))  int i32x4;

// LDS address-space pointer type (for deriving the LDS byte offset for TDM)
typedef __attribute__((address_space(3))) __bf16 lds_bf16;

// ---- feature probes (compile-safe: fall back if the builtin is absent) ----
#if __has_builtin(__builtin_amdgcn_global_load_async_to_lds_b128)
#define USE_ASYNC 1
#else
#define USE_ASYNC 0
#endif
#if __has_builtin(__builtin_amdgcn_tensor_load_to_lds)
#define USE_TDM 1
#else
#define USE_TDM 0
#endif

__device__ inline void wait_async_all() {
#if __has_builtin(__builtin_amdgcn_s_wait_asynccnt)
  __builtin_amdgcn_s_wait_asynccnt(0);
#else
  asm volatile("s_wait_asynccnt 0" ::: "memory");
#endif
}

// ---------------- workspace layout (bytes) ----------------
static const size_t SZ_XB  = (size_t)B_*N_*C_*2;          // x as bf16
static const size_t SZ_WF  = (size_t)C_*QKV_*2;           // fused [Wq|Wkv]^T bf16
static const size_t SZ_PW  = (size_t)C_*C_*2;             // proj_w^T bf16
static const size_t SZ_QKV = (size_t)B_*N_*QKV_*2;        // qkv bf16
static const size_t SZ_A   = (size_t)B_*AG_*C_*4;         // agent tokens f32
static const size_t SZ_PB  = (size_t)HEADS_*AG_*N_*4;     // agent-side bias f32
static const size_t SZ_AB  = (size_t)HEADS_*N_*AG_*4;     // query-side bias f32
static const size_t SZ_AV  = (size_t)B_*HEADS_*AG_*HD_*4; // agent_v f32
static const size_t SZ_OB  = (size_t)B_*N_*C_*2;          // attn-out (+conv) bf16

static const size_t OFF_XB  = 0;
static const size_t OFF_WF  = OFF_XB  + SZ_XB;
static const size_t OFF_PW  = OFF_WF  + SZ_WF;
static const size_t OFF_QKV = OFF_PW  + SZ_PW;
static const size_t OFF_A   = OFF_QKV + SZ_QKV;
static const size_t OFF_PB  = OFF_A   + SZ_A;
static const size_t OFF_AB  = OFF_PB  + SZ_PB;
static const size_t OFF_AV  = OFF_AB  + SZ_AB;
static const size_t OFF_OB  = OFF_AV  + SZ_AV;
static const size_t WS_NEED = OFF_OB  + SZ_OB;   // ~305 MB

// ---------------- small helper kernels ----------------
__global__ __launch_bounds__(256) void k_cast(const float* __restrict__ in,
                                              bf16* __restrict__ out, size_t n) {
  size_t i = (size_t)blockIdx.x * 256 + threadIdx.x;
  if (i < n) out[i] = (bf16)in[i];
}

// fused [Wq | Wkv] stored TRANSPOSED: wf_t[n][k], n in [0,1728), k in [0,576)
__global__ __launch_bounds__(256) void k_build_wf_t(const float* __restrict__ Wq,
                                                    const float* __restrict__ Wkv,
                                                    bf16* __restrict__ wft) {
  int i = blockIdx.x * 256 + threadIdx.x;
  if (i >= QKV_ * C_) return;
  int n = i / C_, k = i % C_;
  float v = (n < C_) ? Wq[k * C_ + n] : Wkv[k * (2 * C_) + (n - C_)];
  wft[i] = (bf16)v;
}

// generic transpose-cast: out[n][k] = W[k][n], W is [K][Nc]
__global__ __launch_bounds__(256) void k_transpose_cast(const float* __restrict__ W,
                                                        bf16* __restrict__ out,
                                                        int K, int Nc) {
  int i = blockIdx.x * 256 + threadIdx.x;
  if (i >= K * Nc) return;
  int n = i / K, k = i % K;
  out[i] = (bf16)W[k * Nc + n];
}

// stage 32 contiguous bytes global -> LDS (async if available)
__device__ inline void stage32(const bf16* __restrict__ src, bf16* dst) {
#if USE_ASYNC
  __builtin_amdgcn_global_load_async_to_lds_b128(
      (i32x4*)src,       (i32x4*)dst,       0, 0);
  __builtin_amdgcn_global_load_async_to_lds_b128(
      (i32x4*)(src + 8), (i32x4*)(dst + 8), 0, 0);
#else
  uint4 d0 = *(const uint4*)(src);
  uint4 d1 = *(const uint4*)(src + 8);
  *(uint4*)(dst)     = d0;
  *(uint4*)(dst + 8) = d1;
#endif
}

// ---------------- WMMA GEMM: C[M,N] = A[M,K] @ Bt[N,K]^T (bf16, f32 acc) ----
// Block tile 128x128x32, 256 threads = 8 wave32 waves (2 along M, 4 along N),
// each wave owns 4x2 = 8 v_wmma_f32_16x16x32_bf16 accumulators. Both LDS
// slabs are K-contiguous per row/col so every fragment is two ds_load_b128.
__global__ __launch_bounds__(256) void k_gemm_wmma(
    const bf16* __restrict__ A, const bf16* __restrict__ Bt,
    bf16* __restrict__ Cb, float* __restrict__ Cf,
    const float* __restrict__ bias, int M, int Nn, int K)
{
  __shared__ bf16 As[128 * 32];   // [row][k]
  __shared__ bf16 Bs[128 * 32];   // [col][k]

  const int tid   = threadIdx.x;
  const int lane  = tid & 31;
  const int wave  = tid >> 5;
  const int waveM = wave & 1;   // 0..1  -> 64 rows each
  const int waveN = wave >> 1;  // 0..3  -> 32 cols each
  const int l15   = lane & 15;
  const int lh    = lane >> 4;

  const int blockN0 = blockIdx.x * 128;
  const int blockM0 = blockIdx.y * 128;

  f32x8 acc[4][2];
  const f32x8 zero = {0.f,0.f,0.f,0.f,0.f,0.f,0.f,0.f};
#pragma unroll
  for (int tm = 0; tm < 4; ++tm)
#pragma unroll
    for (int tn = 0; tn < 2; ++tn) acc[tm][tn] = zero;

  const int sRow = tid >> 1, sSeg = tid & 1;   // 128 rows/cols x 2 x (16 bf16)
  const bool bOk = (blockN0 + sRow) < Nn;

  for (int k0 = 0; k0 < K; k0 += 32) {
    { // ---- stage A slab: 128 rows x 32 k ----
      const bf16* src = A + (size_t)(blockM0 + sRow) * K + k0 + sSeg * 16;
      stage32(src, &As[sRow * 32 + sSeg * 16]);
      if (k0 + 32 < K) __builtin_prefetch(src + 32, 0, 1);  // global_prefetch_b8
    }
    { // ---- stage B slab: 128 cols x 32 k (transposed weights) ----
      if (bOk) {
        const bf16* src = Bt + (size_t)(blockN0 + sRow) * K + k0 + sSeg * 16;
        stage32(src, &Bs[sRow * 32 + sSeg * 16]);
        if (k0 + 32 < K) __builtin_prefetch(src + 32, 0, 1);
      } else {
        uint4 z = {0, 0, 0, 0};
        *(uint4*)(&Bs[sRow * 32 + sSeg * 16])     = z;
        *(uint4*)(&Bs[sRow * 32 + sSeg * 16 + 8]) = z;
      }
    }
#if USE_ASYNC
    wait_async_all();
#endif
    __syncthreads();

    // ---- B fragments: 32x16, lanes 0-15 N=lane hold K 0..15, 16-31 K 16..31
    bf16x16 bfrag[2];
#pragma unroll
    for (int tn = 0; tn < 2; ++tn) {
      int col = waveN * 32 + tn * 16 + l15;
#pragma unroll
      for (int j = 0; j < 16; ++j)
        bfrag[tn][j] = Bs[col * 32 + lh * 16 + j];
    }
    // ---- A fragments: 16x32, lane row=M, K halves per lane group ----
#pragma unroll
    for (int tm = 0; tm < 4; ++tm) {
      int row = waveM * 64 + tm * 16 + l15;
      bf16x16 afrag;
#pragma unroll
      for (int j = 0; j < 16; ++j)
        afrag[j] = As[row * 32 + lh * 16 + j];
      acc[tm][0] = __builtin_amdgcn_wmma_f32_16x16x32_bf16(
          false, afrag, false, bfrag[0], (short)0, acc[tm][0], false, false);
      acc[tm][1] = __builtin_amdgcn_wmma_f32_16x16x32_bf16(
          false, afrag, false, bfrag[1], (short)0, acc[tm][1], false, false);
    }
    __syncthreads();
  }

  // ---- store C: VGPR i -> row = i + 8*(lane>=16), col = lane&15 ----
#pragma unroll
  for (int tm = 0; tm < 4; ++tm)
#pragma unroll
    for (int tn = 0; tn < 2; ++tn) {
      int col = blockN0 + waveN * 32 + tn * 16 + l15;
      if (col < Nn) {
        int rowBase = blockM0 + waveM * 64 + tm * 16 + lh * 8;
        float bv = bias ? bias[col] : 0.f;
#pragma unroll
        for (int i = 0; i < 8; ++i) {
          size_t o = (size_t)(rowBase + i) * Nn + col;
          float v = acc[tm][tn][i];
          if (Cf) Cf[o] = v + bv;
          else    Cb[o] = (bf16)v;
        }
      }
    }
}

// ---------------- agent-token pooling: 8x8 block means of q ----------------
__global__ __launch_bounds__(256) void k_pool(const bf16* __restrict__ qkv,
                                              float* __restrict__ a_ag) {
  int idx = blockIdx.x * 256 + threadIdx.x;
  if (idx >= B_ * AG_ * C_) return;
  int c = idx % C_;
  int t = idx / C_;
  int ag = t % AG_;
  int b  = t / AG_;
  int p1 = ag / 7, p2 = ag % 7;
  float s = 0.f;
  for (int h2 = 0; h2 < 8; ++h2)
    for (int w2 = 0; w2 < 8; ++w2) {
      int n = (p1 * 8 + h2) * HW_ + p2 * 8 + w2;
      s += (float)qkv[(size_t)(b * N_ + n) * QKV_ + c];
    }
  a_ag[idx] = s * (1.f / 64.f);
}

// ---------------- bilinear 7x7 -> 56x56, half-pixel centers, edge clamp ----
__device__ inline float bilin7(const float* __restrict__ p, int i, int j) {
  float fy = (i + 0.5f) * 0.125f - 0.5f;
  float fx = (j + 0.5f) * 0.125f - 0.5f;
  int y0 = (int)floorf(fy), x0 = (int)floorf(fx);
  float wy = fy - (float)y0, wx = fx - (float)x0;
  int y1 = y0 + 1, x1 = x0 + 1;
  y0 = y0 < 0 ? 0 : (y0 > 6 ? 6 : y0);
  y1 = y1 < 0 ? 0 : (y1 > 6 ? 6 : y1);
  x0 = x0 < 0 ? 0 : (x0 > 6 ? 6 : x0);
  x1 = x1 < 0 ? 0 : (x1 > 6 ? 6 : x1);
  float v00 = p[y0 * 7 + x0], v01 = p[y0 * 7 + x1];
  float v10 = p[y1 * 7 + x0], v11 = p[y1 * 7 + x1];
  return (1.f - wy) * ((1.f - wx) * v00 + wx * v01) +
         wy * ((1.f - wx) * v10 + wx * v11);
}

__global__ __launch_bounds__(256) void k_bias_agent(
    const float* __restrict__ an, const float* __restrict__ ah,
    const float* __restrict__ aw, float* __restrict__ pb) {
  int idx = blockIdx.x * 256 + threadIdx.x;
  if (idx >= HEADS_ * AG_ * N_) return;
  int n = idx % N_;
  int t = idx / N_;
  int ag = t % AG_;
  int h  = t / AG_;
  int i = n / HW_, j = n % HW_;
  float v = bilin7(an + (h * AG_ + ag) * 49, i, j);
  v += ah[(h * AG_ + ag) * HW_ + i] + aw[(h * AG_ + ag) * HW_ + j];
  pb[idx] = v;  // [h][ag][n]
}

__global__ __launch_bounds__(256) void k_bias_query(
    const float* __restrict__ na, const float* __restrict__ ha,
    const float* __restrict__ wa, float* __restrict__ ab) {
  int idx = blockIdx.x * 256 + threadIdx.x;
  if (idx >= HEADS_ * N_ * AG_) return;
  int r = idx % AG_;
  int t = idx / AG_;
  int n = t % N_;
  int h = t / N_;
  int i = n / HW_, j = n % HW_;
  float v = bilin7(na + (h * AG_ + r) * 49, i, j);
  v += ha[(h * HW_ + i) * AG_ + r] + wa[(h * HW_ + j) * AG_ + r];
  ab[idx] = v;  // [h][n][ag]
}

// ---------------- agent attention: one WG per (b, h, agent row) ----------
// scores[3136] staged in LDS, block softmax, rank-1 accumulation of v.
// K rows (stride 1728 halves in qkv) are DMAed chunk-wise into LDS by the
// Tensor Data Mover when available (2D tile 72 x CHUNK_, TENSORcnt-waited).
__global__ __launch_bounds__(256) void k_agent_attn(
    const bf16* __restrict__ qkv, const float* __restrict__ a_ag,
    const float* __restrict__ pb, float* __restrict__ agent_v) {
  __shared__ float sc[N_];
  __shared__ float arow[HD_];
  __shared__ float red[8];
  __shared__ float bmax, bsum;
  __shared__ float outAcc[HD_];
#if USE_TDM
  __shared__ bf16 kbuf[CHUNK_ * HD_];
#endif

  const int blk = blockIdx.x;
  const int r = blk % AG_;
  const int h = (blk / AG_) % HEADS_;
  const int b = blk / (AG_ * HEADS_);
  const int tid = threadIdx.x, lane = tid & 31, wv = tid >> 5;
  const float scale = rsqrtf((float)HD_);

  if (tid < HD_) {
    arow[tid]   = a_ag[(size_t)(b * AG_ + r) * C_ + h * HD_ + tid] * scale;
    outAcc[tid] = 0.f;
  }
  __syncthreads();

  float lmax = -1e30f;
#if USE_TDM
  for (int cs = 0; cs < N_; cs += CHUNK_) {
    const int rows = (N_ - cs < CHUNK_) ? (N_ - cs) : CHUNK_;
    if (tid < 32) {  // wave 0 issues the DMA (EXEC ignored by TDM)
      unsigned long long ga =
          (unsigned long long)(size_t)(qkv + (size_t)(b * N_ + cs) * QKV_ + C_ + h * HD_);
      unsigned ldsb = (unsigned)(size_t)(lds_bf16*)kbuf;
      u32x4 g0;
      g0[0] = 1u;                                   // count=1, user mode
      g0[1] = ldsb;                                 // lds_addr
      g0[2] = (unsigned)(ga & 0xffffffffu);         // global_addr lo
      g0[3] = (unsigned)((ga >> 32) & 0x01ffffffu)  // global_addr hi (57b)
              | 0x80000000u;                        // type=2 ("image")
      i32x8 g1;
      g1[0] = (int)0x00010000u;                     // data_size=1 (2 bytes)
      g1[1] = (int)((unsigned)HD_ << 16);           // tensor_dim0 = 72
      g1[2] = (int)(((unsigned)rows & 0xffffu) << 16); // tensor_dim1 = rows
      g1[3] = (int)((unsigned)HD_ << 16);           // tile_dim0 = 72
      g1[4] = (int)((unsigned)rows & 0xffffu);      // tile_dim1 = rows
      g1[5] = (int)QKV_;                            // tensor_dim0_stride = 1728
      g1[6] = 0;
      g1[7] = 0;
      i32x4 g2 = {0, 0, 0, 0};
      i32x4 g3 = {0, 0, 0, 0};
      i32x8 g4 = {0, 0, 0, 0, 0, 0, 0, 0};
      __builtin_amdgcn_tensor_load_to_lds(g0, g1, g2, g3, g4, 0);
      __builtin_amdgcn_s_wait_tensorcnt(0);
    }
    __syncthreads();
    if (tid < rows) {
      int n = cs + tid;
      float d = 0.f;
#pragma unroll
      for (int j = 0; j < HD_; ++j) d += arow[j] * (float)kbuf[tid * HD_ + j];
      d += pb[(size_t)(h * AG_ + r) * N_ + n];
      sc[n] = d;
      lmax = fmaxf(lmax, d);
    }
    __syncthreads();  // kbuf reused next chunk
  }
#else
  for (int n = tid; n < N_; n += 256) {
    const bf16* kp = qkv + (size_t)(b * N_ + n) * QKV_ + C_ + h * HD_;
    float d = 0.f;
#pragma unroll
    for (int j = 0; j < HD_; ++j) d += arow[j] * (float)kp[j];
    d += pb[(size_t)(h * AG_ + r) * N_ + n];
    sc[n] = d;
    lmax = fmaxf(lmax, d);
  }
#endif

  for (int o = 16; o > 0; o >>= 1) lmax = fmaxf(lmax, __shfl_xor(lmax, o, 32));
  if (lane == 0) red[wv] = lmax;
  __syncthreads();
  if (tid == 0) {
    float m = red[0];
    for (int i = 1; i < 8; ++i) m = fmaxf(m, red[i]);
    bmax = m;
  }
  __syncthreads();
  const float m = bmax;

  float lsum = 0.f;
  for (int n = tid; n < N_; n += 256) {
    float e = __expf(sc[n] - m);
    sc[n] = e;
    lsum += e;
  }
  for (int o = 16; o > 0; o >>= 1) lsum += __shfl_xor(lsum, o, 32);
  if (lane == 0) red[wv] = lsum;
  __syncthreads();
  if (tid == 0) {
    float s = 0.f;
    for (int i = 0; i < 8; ++i) s += red[i];
    bsum = s;
  }
  __syncthreads();
  const float inv = 1.f / bsum;

  float lacc[HD_];
#pragma unroll
  for (int j = 0; j < HD_; ++j) lacc[j] = 0.f;
  for (int n = tid; n < N_; n += 256) {
    float p = sc[n] * inv;
    const bf16* vp = qkv + (size_t)(b * N_ + n) * QKV_ + 2 * C_ + h * HD_;
#pragma unroll
    for (int j = 0; j < HD_; ++j) lacc[j] += p * (float)vp[j];
  }
#pragma unroll
  for (int j = 0; j < HD_; ++j) atomicAdd(&outAcc[j], lacc[j]);
  __syncthreads();
  if (tid < HD_) agent_v[(size_t)blk * HD_ + tid] = outAcc[tid];  // [b][h][r][d]
}

// ---------------- query attention: thread per (b,h,n), 49-way softmax ------
__global__ __launch_bounds__(256) void k_query_attn(
    const bf16* __restrict__ qkv, const float* __restrict__ a_ag,
    const float* __restrict__ agent_v, const float* __restrict__ ab,
    bf16* __restrict__ ob) {
  __shared__ float aS[AG_ * HD_];
  __shared__ float avS[AG_ * HD_];
  __shared__ float sS[256 * AG_];   // per-thread score row (avoid scratch)

  const int bh = blockIdx.y;
  const int h = bh % HEADS_;
  const int b = bh / HEADS_;
  const int tid = threadIdx.x;

  for (int i = tid; i < AG_ * HD_; i += 256) {
    int r = i / HD_, d = i % HD_;
    aS[i]  = a_ag[(size_t)(b * AG_ + r) * C_ + h * HD_ + d];
    avS[i] = agent_v[((size_t)bh * AG_ + r) * HD_ + d];
  }
  __syncthreads();

  const int n = blockIdx.x * 256 + tid;
  if (n >= N_) return;

  const float scale = rsqrtf((float)HD_);
  float qv[HD_];
  const bf16* qp = qkv + (size_t)(b * N_ + n) * QKV_ + h * HD_;
#pragma unroll
  for (int j = 0; j < HD_; ++j) qv[j] = scale * (float)qp[j];

  float* s = &sS[tid * AG_];
  float m = -1e30f;
  for (int r = 0; r < AG_; ++r) {
    float d = 0.f;
#pragma unroll
    for (int j = 0; j < HD_; ++j) d += qv[j] * aS[r * HD_ + j];
    d += ab[((size_t)h * N_ + n) * AG_ + r];
    s[r] = d;
    m = fmaxf(m, d);
  }
  float sum = 0.f;
  for (int r = 0; r < AG_; ++r) {
    float e = __expf(s[r] - m);
    s[r] = e;
    sum += e;
  }
  const float inv = 1.f / sum;

  bf16* op = ob + (size_t)(b * N_ + n) * C_ + h * HD_;
#pragma unroll
  for (int j = 0; j < HD_; ++j) {
    float o = 0.f;
    for (int r = 0; r < AG_; ++r) o += s[r] * avS[r * HD_ + j];
    op[j] = (bf16)(o * inv);
  }
}

// ---------------- depthwise 3x3 conv on v + scale/bias + relu, fused add ---
__global__ __launch_bounds__(256) void k_conv_add(
    const bf16* __restrict__ qkv, const float* __restrict__ w,
    const float* __restrict__ gamma, const float* __restrict__ beta,
    bf16* __restrict__ ob) {
  size_t idx = (size_t)blockIdx.x * 256 + threadIdx.x;
  if (idx >= (size_t)B_ * C_ * N_) return;
  int n = (int)(idx % N_);
  size_t t = idx / N_;
  int c = (int)(t % C_);
  int b = (int)(t / C_);
  int i = n / HW_, j = n % HW_;
  float acc = 0.f;
#pragma unroll
  for (int di = -1; di <= 1; ++di)
#pragma unroll
    for (int dj = -1; dj <= 1; ++dj) {
      int ii = i + di, jj = j + dj;
      if (ii < 0 || ii >= HW_ || jj < 0 || jj >= HW_) continue;
      float vv = (float)qkv[(size_t)(b * N_ + ii * HW_ + jj) * QKV_ + 2 * C_ + c];
      acc += vv * w[c * 9 + (di + 1) * 3 + (dj + 1)];
    }
  float y = fmaxf(acc * gamma[c] + beta[c], 0.f);
  size_t oi = (size_t)(b * N_ + n) * C_ + c;
  ob[oi] = (bf16)((float)ob[oi] + y);
}

// ---------------- host side ----------------
static inline unsigned cdiv(size_t a, size_t b) { return (unsigned)((a + b - 1) / b); }

extern "C" void kernel_launch(void* const* d_in, const int* in_sizes, int n_in,
                              void* d_out, int out_size, void* d_ws, size_t ws_size,
                              hipStream_t stream) {
  (void)in_sizes; (void)n_in; (void)out_size;
  if (ws_size < WS_NEED) return;  // insufficient scratch — bail deterministically

  const float* x        = (const float*)d_in[0];
  const float* Wq       = (const float*)d_in[1];
  const float* Wkv      = (const float*)d_in[2];
  const float* an_bias  = (const float*)d_in[3];
  const float* na_bias  = (const float*)d_in[4];
  const float* ah_bias  = (const float*)d_in[5];
  const float* aw_bias  = (const float*)d_in[6];
  const float* ha_bias  = (const float*)d_in[7];
  const float* wa_bias  = (const float*)d_in[8];
  const float* dwc_w    = (const float*)d_in[9];
  const float* dwc_g    = (const float*)d_in[10];
  const float* dwc_b    = (const float*)d_in[11];
  const float* proj_w   = (const float*)d_in[12];
  const float* proj_b   = (const float*)d_in[13];

  char* ws = (char*)d_ws;
  bf16*  xb   = (bf16*) (ws + OFF_XB);
  bf16*  wft  = (bf16*) (ws + OFF_WF);   // [1728][576]
  bf16*  pwt  = (bf16*) (ws + OFF_PW);   // [576][576] (proj_w^T)
  bf16*  qkv  = (bf16*) (ws + OFF_QKV);
  float* a_ag = (float*)(ws + OFF_A);
  float* pb   = (float*)(ws + OFF_PB);
  float* ab   = (float*)(ws + OFF_AB);
  float* av   = (float*)(ws + OFF_AV);
  bf16*  ob   = (bf16*) (ws + OFF_OB);
  float* out  = (float*)d_out;

  const size_t nx = (size_t)B_ * N_ * C_;
  const int M = B_ * N_;  // 50176, multiple of 128

  // 1) casts / weight fusion (weights stored transposed for the GEMM B path)
  k_cast<<<cdiv(nx, 256), 256, 0, stream>>>(x, xb, nx);
  k_build_wf_t<<<cdiv((size_t)QKV_ * C_, 256), 256, 0, stream>>>(Wq, Wkv, wft);
  k_transpose_cast<<<cdiv((size_t)C_ * C_, 256), 256, 0, stream>>>(proj_w, pwt, C_, C_);

  // 2) fused qkv projection: [50176 x 576] @ [576 x 1728] -> bf16
  {
    dim3 grid(cdiv(QKV_, 128), M / 128);
    k_gemm_wmma<<<grid, 256, 0, stream>>>(xb, wft, qkv, nullptr, nullptr,
                                          M, QKV_, C_);
  }

  // 3) agent tokens (8x8 mean pooling of q)
  k_pool<<<cdiv((size_t)B_ * AG_ * C_, 256), 256, 0, stream>>>(qkv, a_ag);

  // 4) positional biases (bilinear upsample + separable components)
  k_bias_agent<<<cdiv((size_t)HEADS_ * AG_ * N_, 256), 256, 0, stream>>>(
      an_bias, ah_bias, aw_bias, pb);
  k_bias_query<<<cdiv((size_t)HEADS_ * N_ * AG_, 256), 256, 0, stream>>>(
      na_bias, ha_bias, wa_bias, ab);

  // 5) agent attention (softmax over N fused in LDS, TDM-staged K when avail)
  k_agent_attn<<<B_ * HEADS_ * AG_, 256, 0, stream>>>(qkv, a_ag, pb, av);

  // 6) query attention (softmax over 49 agents), writes attn-out bf16
  {
    dim3 grid(cdiv(N_, 256), B_ * HEADS_);
    k_query_attn<<<grid, 256, 0, stream>>>(qkv, a_ag, av, ab, ob);
  }

  // 7) depthwise conv branch fused into attn-out
  k_conv_add<<<cdiv((size_t)B_ * C_ * N_, 256), 256, 0, stream>>>(
      qkv, dwc_w, dwc_g, dwc_b, ob);

  // 8) output projection + bias -> d_out (fp32)
  {
    dim3 grid(cdiv(C_, 128), M / 128);
    k_gemm_wmma<<<grid, 256, 0, stream>>>(ob, pwt, nullptr, out, proj_b,
                                          M, C_, C_);
  }
}
